// BiLSTM_CRF_56753697849811
// MI455X (gfx1250) — matrix operand order repaired and checked by reading, not compiled, not run on hardware
//
#include <hip/hip_runtime.h>
#include <hip/hip_bf16.h>

// ---------------------------------------------------------------------------
// BiLSTM-CRF forward loss for gfx1250 (MI455X), wave32 + WMMA bf16.
// B=64, T=512, E=256, HD=256, H=512, K=32, V=50000.
// ---------------------------------------------------------------------------

typedef __attribute__((ext_vector_type(16))) __bf16 v16bf;
typedef __attribute__((ext_vector_type(8)))  __bf16 v8bf;
typedef __attribute__((ext_vector_type(8)))  float  v8f;

#define TSEQ 512
#define EDIM 256
#define HD   256
#define HDIM 512
#define KTAG 32

__device__ __forceinline__ float sigf(float x) { return 1.0f / (1.0f + __expf(-x)); }

__device__ __forceinline__ v8f wmma_bf16(v16bf a, v16bf b, v8f c) {
    // D = A(16x32 bf16) * B(32x16 bf16) + C(16x16 f32)
    return __builtin_amdgcn_wmma_f32_16x16x32_bf16(false, a, false, b, (short)0, c,
                                                   false, false);
}

// Build a 16x32 bf16 A-fragment for lane `lane` from a row-major bf16 LDS
// buffer (row length 512 elems). Layout: lane m (0..15) row m,
// K = hi*16+{0..15} contiguous (matches packed B tiles).
__device__ __forceinline__ v16bf a_frag(const __bf16* row, int kbase, int hi) {
    v8bf lo = *(const v8bf*)(row + kbase + hi * 16);
    v8bf hg = *(const v8bf*)(row + kbase + hi * 16 + 8);
    return __builtin_shufflevector(lo, hg, 0, 1, 2, 3, 4, 5, 6, 7,
                                   8, 9, 10, 11, 12, 13, 14, 15);
}

// ---------------------------------------------------------------------------
// Weight packing into WMMA B-fragment tiles.
// Tile (kt, nt) holds B[k = kt*32 + kk, n = nt*16 + (lane&15)], with
// kk = (lane>>4)*16 + e ; storage = tile*512 + lane*16 + e  (bf16).
// ---------------------------------------------------------------------------

// Recurrence weights: combined [w_ih (g,e) | w_hh (g,k)] -> (1024 x 512),
// 16 kt x 64 nt tiles. One call per direction. 524288 elems -> 2048 blocks.
__global__ __launch_bounds__(256) void pack_rec_w(const float* __restrict__ w_ih,
                                                  const float* __restrict__ w_hh,
                                                  __bf16* __restrict__ out) {
    int idx    = blockIdx.x * 256 + threadIdx.x;
    int tile   = idx >> 9;
    int within = idx & 511;
    int lane   = within >> 4;
    int e      = within & 15;
    int kt = tile >> 6;
    int nt = tile & 63;
    int g    = nt * 16 + (lane & 15);
    int kidx = kt * 32 + (lane >> 4) * 16 + e;
    float v = (kidx < EDIM) ? w_ih[(size_t)g * EDIM + kidx]
                            : w_hh[(size_t)g * HD + (kidx - EDIM)];
    out[idx] = (__bf16)v;
}

// Generic (N x K) row-major fp32 weight -> bf16 fragment tiles, K multiple of 32.
__global__ __launch_bounds__(256) void pack_mlp_w(const float* __restrict__ W,
                                                  __bf16* __restrict__ out,
                                                  int NT, int K) {
    int idx    = blockIdx.x * 256 + threadIdx.x;
    int tile   = idx >> 9;
    int within = idx & 511;
    int lane   = within >> 4;
    int e      = within & 15;
    int kt = tile / NT;
    int nt = tile % NT;
    int n    = nt * 16 + (lane & 15);
    int kidx = kt * 32 + (lane >> 4) * 16 + e;
    out[idx] = (__bf16)W[(size_t)n * K + kidx];
}

// Combined LSTM bias per direction: brec[dir*1024 + g] = b_ih[g] + b_hh[g].
__global__ __launch_bounds__(256) void prep_bias(const float* __restrict__ bif,
                                                 const float* __restrict__ bhf,
                                                 const float* __restrict__ bib,
                                                 const float* __restrict__ bhb,
                                                 float* __restrict__ out) {
    int i = blockIdx.x * 256 + threadIdx.x;  // < 2048
    if (i < 1024) out[i] = bif[i] + bhf[i];
    else          out[i] = bib[i - 1024] + bhb[i - 1024];
}

// ---------------------------------------------------------------------------
// Bidirectional LSTM recurrence, input projection fused.
// grid = 8 blocks: dir = blockIdx.x>>2, batch tile (16 rows) = blockIdx.x&3.
// Per step: gates(16x1024) = [x_t | h] (16x512 bf16) @ Wrec^T via WMMA.
// Wave w owns gate tiles nt = w + 8*s, s=0..7 => i/f/g/o of a hidden unit land
// in the same lane/element of acc[s], acc[s+2], acc[s+4], acc[s+6] -> the cell
// update is register-local; c lives in VGPRs; h goes back to LDS for t+1.
// ---------------------------------------------------------------------------
__global__ __launch_bounds__(256) void bilstm_rec(const int* __restrict__ ids,
                                                  const float* __restrict__ emb,
                                                  const __bf16* __restrict__ wrec,
                                                  const float* __restrict__ brec,
                                                  __bf16* __restrict__ lstm_out) {
    const int dir   = blockIdx.x >> 2;
    const int btile = blockIdx.x & 3;
    const int tid   = threadIdx.x;
    const int wave  = tid >> 5;
    const int lane  = tid & 31;
    const int n16   = lane & 15;
    const int hi    = lane >> 4;

    __shared__ alignas(16) __bf16 As[16][512];  // [batch row][ x(0:256) | h(256:512) ]

    // h0 = 0
    for (int idx = tid; idx < 16 * HD; idx += 256)
        As[idx >> 8][EDIM + (idx & 255)] = (__bf16)0.0f;

    v8f creg[2];
#pragma unroll
    for (int h2 = 0; h2 < 2; ++h2)
#pragma unroll
        for (int r = 0; r < 8; ++r) creg[h2][r] = 0.0f;

    float bv[8];
#pragma unroll
    for (int s = 0; s < 8; ++s)
        bv[s] = brec[dir * 1024 + (wave + 8 * s) * 16 + n16];

    const __bf16* wdir = wrec + (size_t)dir * (16 * 64 * 512) + (size_t)lane * 16;

    for (int it = 0; it < TSEQ; ++it) {
        const int t = dir ? (TSEQ - 1 - it) : it;

        // Stage x_t = emb[ids[b,t]] as bf16 into LDS (coalesced over e).
        for (int idx = tid; idx < 16 * EDIM; idx += 256) {
            int b = idx >> 8, e = idx & 255;
            int id = ids[(size_t)(btile * 16 + b) * TSEQ + t];
            As[b][e] = (__bf16)emb[(size_t)id * EDIM + e];
        }
        __syncthreads();

        v8f acc[8];
#pragma unroll
        for (int s = 0; s < 8; ++s)
#pragma unroll
            for (int r = 0; r < 8; ++r) acc[s][r] = bv[s];

        for (int kt = 0; kt < 16; ++kt) {
            v16bf a = a_frag(&As[n16][0], kt * 32, hi);
            const __bf16* wk = wdir + (size_t)kt * (64 * 512);
#pragma unroll
            for (int s = 0; s < 8; ++s) {
                v16bf b = *(const v16bf*)(wk + (size_t)(wave + 8 * s) * 512);
                acc[s] = wmma_bf16(a, b, acc[s]);
            }
        }
        __syncthreads();  // all WMMA reads of As done before h writes

        // LSTM cell update (register-local gates), write h for next step.
#pragma unroll
        for (int h2 = 0; h2 < 2; ++h2) {
            const int j = (wave + 8 * h2) * 16 + n16;  // hidden unit 0..255
#pragma unroll
            for (int r = 0; r < 8; ++r) {
                float iv = acc[0 + h2][r];
                float fv = acc[2 + h2][r];
                float gv = acc[4 + h2][r];
                float ov = acc[6 + h2][r];
                float c  = sigf(fv) * creg[h2][r] + sigf(iv) * tanhf(gv);
                float hv = sigf(ov) * tanhf(c);
                creg[h2][r] = c;
                const int m = hi * 8 + r;  // batch row in tile
                As[m][EDIM + j] = (__bf16)hv;
                lstm_out[((size_t)(btile * 16 + m) * TSEQ + t) * HDIM +
                         dir * HD + j] = (__bf16)hv;
            }
        }
        // x-region (next iter) and h-region are disjoint; next __syncthreads()
        // at top of loop orders h writes before the next WMMA phase.
    }
}

// ---------------------------------------------------------------------------
// Generic bf16 GEMM + bias (+ReLU): out(M x Nout) = in(M x 512) @ W^T.
// One block = 16 rows of M; 8 waves split NT column tiles (<=4 each).
// ---------------------------------------------------------------------------
__global__ __launch_bounds__(256) void mlp_gemm(const __bf16* __restrict__ in,
                                                const __bf16* __restrict__ wpk,
                                                const float* __restrict__ bias,
                                                const int NT, const int relu,
                                                __bf16* __restrict__ out_bf,
                                                float* __restrict__ out_f,
                                                const int Nout) {
    const int tid  = threadIdx.x;
    const int wave = tid >> 5;
    const int lane = tid & 31;
    const int n16  = lane & 15;
    const int hi   = lane >> 4;

    __shared__ alignas(16) __bf16 Ain[16 * 512];

    // 16 input rows are contiguous (row-major, K=512): flat 16B copies.
    const uint4* src = (const uint4*)(in + (size_t)blockIdx.x * (16 * 512));
    uint4* dst = (uint4*)Ain;
#pragma unroll
    for (int i = 0; i < 4; ++i) dst[tid + 256 * i] = src[tid + 256 * i];
    __syncthreads();

    int nts[4];
    int cnt = 0;
    for (int nt = wave; nt < NT; nt += 8)
        if (cnt < 4) nts[cnt++] = nt;

    v8f acc[4];
    for (int i = 0; i < cnt; ++i) {
        float b = bias[nts[i] * 16 + n16];
#pragma unroll
        for (int r = 0; r < 8; ++r) acc[i][r] = b;
    }

    for (int kt = 0; kt < 16; ++kt) {
        v16bf a = a_frag(&Ain[n16 * 512], kt * 32, hi);
        for (int i = 0; i < cnt; ++i) {
            const __bf16* bp = wpk + ((size_t)(kt * NT + nts[i])) * 512 + lane * 16;
            if (kt < 15) __builtin_prefetch(bp + (size_t)NT * 512, 0, 0);
            v16bf b = *(const v16bf*)bp;
            acc[i] = wmma_bf16(a, b, acc[i]);
        }
    }

    const size_t row0 = (size_t)blockIdx.x * 16;
    for (int i = 0; i < cnt; ++i) {
        const int ncol = nts[i] * 16 + n16;
#pragma unroll
        for (int r = 0; r < 8; ++r) {
            float v = acc[i][r];
            if (relu) v = fmaxf(v, 0.0f);
            size_t off = (row0 + hi * 8 + r) * (size_t)Nout + ncol;
            if (out_bf) out_bf[off] = (__bf16)v;
            else        out_f[off] = v;
        }
    }
}

// ---------------------------------------------------------------------------
// CRF NLL. grid=2 blocks x 1024 threads; wave w = batch (32/block), lane = tag.
// Alpha recursion in LDS; denom + gold score via wave32 shuffles.
// ---------------------------------------------------------------------------
__global__ __launch_bounds__(1024) void crf_nll(const float* __restrict__ emis,
                                                const int* __restrict__ tags,
                                                const unsigned char* __restrict__ mask,
                                                const float* __restrict__ start_t,
                                                const float* __restrict__ end_t,
                                                const float* __restrict__ trans,
                                                float* __restrict__ out) {
    const int tid = threadIdx.x;
    const int bl  = tid >> 5;   // batch within block == wave id
    const int k   = tid & 31;   // tag
    const int gb  = blockIdx.x * 32 + bl;

    __shared__ float tr[32][33];
    __shared__ float al[32][33];

    tr[bl][k] = trans[bl * 32 + k];  // 1024 threads cover 32x32 exactly
    const size_t ebase = (size_t)gb * TSEQ * KTAG;
    al[bl][k] = start_t[k] + emis[ebase + k];
    __syncthreads();

    for (int t = 1; t < TSEQ; ++t) {
        float em = emis[ebase + (size_t)t * KTAG + k];
        int mt = mask[(size_t)gb * TSEQ + t];
        float mx = -3.0e38f;
#pragma unroll 4
        for (int j = 0; j < 32; ++j) mx = fmaxf(mx, al[bl][j] + tr[j][k]);
        float s = 0.0f;
#pragma unroll 4
        for (int j = 0; j < 32; ++j) s += __expf(al[bl][j] + tr[j][k] - mx);
        float nxt = mx + __logf(s) + em;
        __syncthreads();
        if (mt) al[bl][k] = nxt;
        __syncthreads();
    }

    // denom = logsumexp_k(alpha + end)
    float v = al[bl][k] + end_t[k];
    float mx = v;
    for (int o = 16; o > 0; o >>= 1) mx = fmaxf(mx, __shfl_xor(mx, o, 32));
    float s = __expf(v - mx);
    for (int o = 16; o > 0; o >>= 1) s += __shfl_xor(s, o, 32);
    float denom = mx + __logf(s);

    // gold-path score, lanes stride over time
    float partial = 0.0f;
    int cnt = 0;
    for (int t = k; t < TSEQ; t += 32) {
        int tg = tags[(size_t)gb * TSEQ + t];
        int m  = mask[(size_t)gb * TSEQ + t];
        cnt += m;
        float e = emis[ebase + (size_t)t * KTAG + tg];
        if (t == 0) {
            partial += start_t[tg] + e;
        } else {
            int tp = tags[(size_t)gb * TSEQ + t - 1];
            partial += (tr[tp][tg] + e) * (m ? 1.0f : 0.0f);
        }
    }
    for (int o = 16; o > 0; o >>= 1) {
        partial += __shfl_xor(partial, o, 32);
        cnt     += __shfl_xor(cnt, o, 32);
    }
    if (k == 0) {
        int last = tags[(size_t)gb * TSEQ + (cnt - 1)];
        float score = partial + end_t[last];
        atomicAdd(out, -(score - denom) * (1.0f / 64.0f));
    }
}

// ---------------------------------------------------------------------------
// Host: workspace layout + launch sequence (all on `stream`, capture-safe).
// ---------------------------------------------------------------------------
static const size_t OFF_WREC = 0;                              // 2 MB (2 dirs)
static const size_t OFF_BREC = 2097152;                        // 8 KB
static const size_t OFF_W1P  = 2105344;                        // 512 KB
static const size_t OFF_W2P  = 2629632;                        // 512 KB
static const size_t OFF_WFP  = 3153920;                        // 32 KB
static const size_t OFF_LSTM = 3186688;                        // 32 MB bf16
static const size_t OFF_H1   = 36741120;                       // 32 MB bf16
static const size_t OFF_H2   = 70295552;                       // 32 MB bf16
static const size_t OFF_EMIS = 103849984;                      // 4 MB f32

extern "C" void kernel_launch(void* const* d_in, const int* in_sizes, int n_in,
                              void* d_out, int out_size, void* d_ws, size_t ws_size,
                              hipStream_t stream) {
    const int*           ids     = (const int*)d_in[0];
    const int*           labels  = (const int*)d_in[1];
    const unsigned char* mask    = (const unsigned char*)d_in[2];
    const float*         emb     = (const float*)d_in[3];
    const float*         w_ih_f  = (const float*)d_in[4];
    const float*         w_hh_f  = (const float*)d_in[5];
    const float*         b_ih_f  = (const float*)d_in[6];
    const float*         b_hh_f  = (const float*)d_in[7];
    const float*         w_ih_b  = (const float*)d_in[8];
    const float*         w_hh_b  = (const float*)d_in[9];
    const float*         b_ih_b  = (const float*)d_in[10];
    const float*         b_hh_b  = (const float*)d_in[11];
    const float*         W1      = (const float*)d_in[12];
    const float*         b1      = (const float*)d_in[13];
    const float*         W2      = (const float*)d_in[14];
    const float*         b2      = (const float*)d_in[15];
    const float*         Wf      = (const float*)d_in[16];
    const float*         bfv     = (const float*)d_in[17];
    const float*         start_t = (const float*)d_in[18];
    const float*         end_t   = (const float*)d_in[19];
    const float*         trans   = (const float*)d_in[20];

    char* ws = (char*)d_ws;
    __bf16* wrec = (__bf16*)(ws + OFF_WREC);
    float*  brec = (float*)(ws + OFF_BREC);
    __bf16* w1p  = (__bf16*)(ws + OFF_W1P);
    __bf16* w2p  = (__bf16*)(ws + OFF_W2P);
    __bf16* wfp  = (__bf16*)(ws + OFF_WFP);
    __bf16* lstm = (__bf16*)(ws + OFF_LSTM);
    __bf16* h1   = (__bf16*)(ws + OFF_H1);
    __bf16* h2   = (__bf16*)(ws + OFF_H2);
    float*  emis = (float*)(ws + OFF_EMIS);

    // Weight packing (parallel, one-shot per call; deterministic).
    pack_rec_w<<<2048, 256, 0, stream>>>(w_ih_f, w_hh_f, wrec);
    pack_rec_w<<<2048, 256, 0, stream>>>(w_ih_b, w_hh_b, wrec + (size_t)16 * 64 * 512);
    prep_bias<<<8, 256, 0, stream>>>(b_ih_f, b_hh_f, b_ih_b, b_hh_b, brec);
    pack_mlp_w<<<1024, 256, 0, stream>>>(W1, w1p, 32, 512);
    pack_mlp_w<<<1024, 256, 0, stream>>>(W2, w2p, 32, 512);
    pack_mlp_w<<<64, 256, 0, stream>>>(Wf, wfp, 2, 512);

    // Fused embedding + input-proj + recurrence, both directions in one grid.
    bilstm_rec<<<8, 256, 0, stream>>>(ids, emb, wrec, brec, lstm);

    // MLP + emissions (M = B*T = 32768 -> 2048 row-tiles of 16).
    mlp_gemm<<<2048, 256, 0, stream>>>(lstm, w1p, b1, 32, 1, h1, nullptr, 512);
    mlp_gemm<<<2048, 256, 0, stream>>>(h1, w2p, b2, 32, 1, h2, nullptr, 512);
    mlp_gemm<<<2048, 256, 0, stream>>>(h2, wfp, bfv, 2, 0, nullptr, emis, 32);

    // CRF NLL -> scalar.
    hipMemsetAsync(d_out, 0, sizeof(float), stream);
    crf_nll<<<2, 1024, 0, stream>>>(emis, labels, mask, start_t, end_t, trans,
                                    (float*)d_out);

    (void)in_sizes; (void)n_in; (void)out_size; (void)ws_size;
}
